// FastQuantizedLinear_56453050138624
// MI455X (gfx1250) — compile-verified, matrix operand order checked by reading, not run on hardware
//
#include <hip/hip_runtime.h>
#include <stdint.h>

// ---------------------------------------------------------------------------
// Quantized linear: out[M,N] = A[M,K] x ((W_int8[N,K] - zp[N]) * scale[N])^T + bias
// M = 8192, N = 4096, K = 4096.
//
// Fast path (needs 96MB workspace):
//   prep kernels:  input f32 -> bf16 ; weight int32 -> dequantized bf16
//   main GEMM:     bf16 WMMA, tiles staged via GLOBAL_LOAD_ASYNC_TO_LDS_B128
//                  3-deep ASYNCcnt pipeline (partial waits), f32 accumulate.
// Fallback (small ws): fused on-the-fly dequant GEMM.
// ---------------------------------------------------------------------------

typedef __attribute__((ext_vector_type(16))) __bf16 v16bf;
typedef __attribute__((ext_vector_type(8)))  __bf16 bf16x8;   // 16 bytes
typedef __attribute__((ext_vector_type(4)))  __bf16 bf16x4;   // 8 bytes
typedef __attribute__((ext_vector_type(8)))  float  v8f;

constexpr int Mtot = 8192;
constexpr int Ntot = 4096;
constexpr int Ktot = 4096;

constexpr int BM = 128;
constexpr int BN = 128;

union V16u { v16bf v; bf16x8 h[2]; };

// ---------------- CDNA5 async copy helpers (gfx1250) ------------------------

__device__ __forceinline__ void async_copy_b128(uint32_t lds_addr, const void* gaddr)
{
    // GLOBAL_LOAD_ASYNC_TO_LDS_B128: LDS[vdst] = MEM[vaddr], 16B per lane,
    // tracked with ASYNCcnt (no VGPR data round-trip).
    asm volatile("global_load_async_to_lds_b128 %0, %1, off"
                 :: "v"(lds_addr), "v"(gaddr)
                 : "memory");
}

__device__ __forceinline__ void wait_asynccnt0()
{
    asm volatile("s_wait_asynccnt 0x0" ::: "memory");
}

// ASYNCcnt completes in order: cnt<=8 with 16 outstanding means the older
// stage (8 instrs) has fully landed while the newest 8 may still be in flight.
__device__ __forceinline__ void wait_asynccnt8()
{
    asm volatile("s_wait_asynccnt 0x8" ::: "memory");
}

// ---------------- prep kernels ---------------------------------------------

__global__ __launch_bounds__(256)
void convert_input_bf16(const float* __restrict__ in, __bf16* __restrict__ out, int n4)
{
    int i = blockIdx.x * blockDim.x + threadIdx.x;
    if (i < n4) {
        float4 v = ((const float4*)in)[i];
        bf16x4 b = { (__bf16)v.x, (__bf16)v.y, (__bf16)v.z, (__bf16)v.w };
        ((bf16x4*)out)[i] = b;
    }
}

__global__ __launch_bounds__(256)
void dequant_weight_bf16(const int* __restrict__ w,
                         const float* __restrict__ scale,
                         const float* __restrict__ zp,
                         __bf16* __restrict__ out, int n4)
{
    int i = blockIdx.x * blockDim.x + threadIdx.x;
    if (i < n4) {
        int4 v = ((const int4*)w)[i];
        int n = (i * 4) >> 12;                 // element / 4096 -> output channel
        float s = scale[n], z = zp[n];
        bf16x4 b = { (__bf16)(((float)v.x - z) * s),
                     (__bf16)(((float)v.y - z) * s),
                     (__bf16)(((float)v.z - z) * s),
                     (__bf16)(((float)v.w - z) * s) };
        ((bf16x4*)out)[i] = b;
    }
}

// ---------------- main GEMM: bf16 tiles via async-to-LDS --------------------

__global__ __launch_bounds__(256)
void qgemm_bf16_async(const __bf16* __restrict__ Abf,
                      const __bf16* __restrict__ Wbf,
                      const float*  __restrict__ bias,
                      float*        __restrict__ out)
{
    constexpr int BK  = 64;        // two wmma K-steps per stage
    constexpr int LST = BK + 8;    // 72 elems = 144B row stride (16B aligned)
    constexpr int NB  = 3;         // 3-deep pipeline

    __shared__ __bf16 Asb[NB][BM][LST];    // 3 x 18KB
    __shared__ __bf16 Wsb[NB][BN][LST];    // 3 x 18KB  -> 108KB total

    const int tid  = threadIdx.x;
    const int bn   = blockIdx.x * BN;
    const int bm   = blockIdx.y * BM;

    const int wave = tid >> 5;
    const int lane = tid & 31;
    const int wm   = (wave >> 2) * 64;
    const int wn   = (wave & 3) * 32;
    const int m16  = lane & 15;
    const int hi   = lane >> 4;

    // copy slots: per tile 128 rows x 64 k = 1024 x 16B chunks; 4 per thread
    int crow[4], ckc[4];
#pragma unroll
    for (int it = 0; it < 4; ++it) {
        int slot = it * 256 + tid;
        crow[it] = slot >> 3;          // 0..127
        ckc[it]  = (slot & 7) * 8;     // 0,8,...,56 (bf16 elems)
    }

    // 8 async-copy instructions per wave per stage (4 A + 4 W)
    auto issue = [&](int kt, int buf) {
        const int k0 = kt * BK;
#pragma unroll
        for (int it = 0; it < 4; ++it) {
            const __bf16* ga = &Abf[(size_t)(bm + crow[it]) * Ktot + k0 + ckc[it]];
            async_copy_b128((uint32_t)(uintptr_t)&Asb[buf][crow[it]][ckc[it]], ga);
            const __bf16* gw = &Wbf[(size_t)(bn + crow[it]) * Ktot + k0 + ckc[it]];
            async_copy_b128((uint32_t)(uintptr_t)&Wsb[buf][crow[it]][ckc[it]], gw);
        }
    };

    v8f acc[4][2] = {};

    auto compute = [&](int buf) {
#pragma unroll
        for (int s = 0; s < 2; ++s) {          // two K=32 steps within the stage
            V16u af[4], bfr[2];
#pragma unroll
            for (int i = 0; i < 4; ++i) {
                const __bf16* p = &Asb[buf][wm + i * 16 + m16][s * 32];
                af[i].h[0] = *(const bf16x8*)&p[hi * 8];
                af[i].h[1] = *(const bf16x8*)&p[16 + hi * 8];
            }
#pragma unroll
            for (int j = 0; j < 2; ++j) {
                const __bf16* p = &Wsb[buf][wn + j * 16 + m16][s * 32];
                bfr[j].h[0] = *(const bf16x8*)&p[hi * 8];
                bfr[j].h[1] = *(const bf16x8*)&p[16 + hi * 8];
            }
#pragma unroll
            for (int i = 0; i < 4; ++i)
#pragma unroll
                for (int j = 0; j < 2; ++j)
                    acc[i][j] = __builtin_amdgcn_wmma_f32_16x16x32_bf16(
                        false, af[i].v, false, bfr[j].v,
                        (short)0, acc[i][j], false, false);
        }
    };

    constexpr int KT = Ktot / BK;   // 64 stages

    // Prime two stages of the pipeline.
    issue(0, 0);
    issue(1, 1);

    int buf = 0;
    for (int kt = 0; kt < KT; ++kt) {
        // Ensure stage kt has landed; stage kt+1's 8 copies may remain in flight.
        if (kt + 1 < KT) wait_asynccnt8();
        else             wait_asynccnt0();
        __syncthreads();               // publish stage kt; readers of buf(kt-1) done

        if (kt + 2 < KT) {
            const int nb = (buf + 2 >= NB) ? (buf + 2 - NB) : (buf + 2);
            issue(kt + 2, nb);         // DMA has two compute stages to complete
        }

        compute(buf);
        buf = (buf + 1 == NB) ? 0 : buf + 1;
    }

#pragma unroll
    for (int i = 0; i < 4; ++i) {
        const int gm0 = bm + wm + i * 16 + hi * 8;
#pragma unroll
        for (int j = 0; j < 2; ++j) {
            const int gn = bn + wn + j * 16 + m16;
            const float bv = bias[gn];
#pragma unroll
            for (int r = 0; r < 8; ++r)
                out[(size_t)(gm0 + r) * Ntot + gn] = acc[i][j][r] + bv;
        }
    }
}

// ---------------- fallback: fused on-the-fly dequant GEMM -------------------

__global__ __launch_bounds__(256)
void qlinear_wmma_bf16(const float* __restrict__ A,
                       const int*   __restrict__ W,
                       const float* __restrict__ scale,
                       const float* __restrict__ zp,
                       const float* __restrict__ bias,
                       float*       __restrict__ out)
{
    constexpr int BK = 32;
    constexpr int LDSS = BK + 8;

    __shared__ __bf16 Asb[2][BM][LDSS];
    __shared__ __bf16 Wsb[2][BN][LDSS];

    const int tid  = threadIdx.x;
    const int bn   = blockIdx.x * BN;
    const int bm   = blockIdx.y * BM;

    const int wave = tid >> 5;
    const int lane = tid & 31;
    const int wm   = (wave >> 2) * 64;
    const int wn   = (wave & 3) * 32;
    const int m16  = lane & 15;
    const int hi   = lane >> 4;

    int   row[4], kc[4];
    float sc[4], zv[4];
#pragma unroll
    for (int it = 0; it < 4; ++it) {
        int slot = it * 256 + tid;
        row[it] = slot >> 3;
        kc[it]  = (slot & 7) * 4;
        sc[it]  = scale[bn + row[it]];
        zv[it]  = zp[bn + row[it]];
    }

    float4 ar[4];
    int4   wr[4];

    auto fetch = [&](int kt) {
        const int k0 = kt * BK;
#pragma unroll
        for (int it = 0; it < 4; ++it) {
            ar[it] = *(const float4*)&A[(size_t)(bm + row[it]) * Ktot + k0 + kc[it]];
            wr[it] = *(const int4*  )&W[(size_t)(bn + row[it]) * Ktot + k0 + kc[it]];
        }
    };

    auto stash = [&](int buf) {
#pragma unroll
        for (int it = 0; it < 4; ++it) {
            bf16x4 av = { (__bf16)ar[it].x, (__bf16)ar[it].y,
                          (__bf16)ar[it].z, (__bf16)ar[it].w };
            *(bf16x4*)&Asb[buf][row[it]][kc[it]] = av;
            const float s = sc[it], z = zv[it];
            bf16x4 wv = { (__bf16)(((float)wr[it].x - z) * s),
                          (__bf16)(((float)wr[it].y - z) * s),
                          (__bf16)(((float)wr[it].z - z) * s),
                          (__bf16)(((float)wr[it].w - z) * s) };
            *(bf16x4*)&Wsb[buf][row[it]][kc[it]] = wv;
        }
    };

    v8f acc[4][2] = {};

    auto compute = [&](int buf) {
        V16u af[4], bfr[2];
#pragma unroll
        for (int i = 0; i < 4; ++i) {
            const __bf16* p = &Asb[buf][wm + i * 16 + m16][0];
            af[i].h[0] = *(const bf16x8*)&p[hi * 8];
            af[i].h[1] = *(const bf16x8*)&p[16 + hi * 8];
        }
#pragma unroll
        for (int j = 0; j < 2; ++j) {
            const __bf16* p = &Wsb[buf][wn + j * 16 + m16][0];
            bfr[j].h[0] = *(const bf16x8*)&p[hi * 8];
            bfr[j].h[1] = *(const bf16x8*)&p[16 + hi * 8];
        }
#pragma unroll
        for (int i = 0; i < 4; ++i)
#pragma unroll
            for (int j = 0; j < 2; ++j)
                acc[i][j] = __builtin_amdgcn_wmma_f32_16x16x32_bf16(
                    false, af[i].v, false, bfr[j].v,
                    (short)0, acc[i][j], false, false);
    };

    constexpr int KT = Ktot / BK;

    fetch(0);
    stash(0);
    __syncthreads();

    for (int kt = 0; kt < KT; ++kt) {
        const int buf = kt & 1;
        if (kt + 1 < KT) fetch(kt + 1);
        compute(buf);
        if (kt + 1 < KT) stash(buf ^ 1);
        __syncthreads();
    }

#pragma unroll
    for (int i = 0; i < 4; ++i) {
        const int gm0 = bm + wm + i * 16 + hi * 8;
#pragma unroll
        for (int j = 0; j < 2; ++j) {
            const int gn = bn + wn + j * 16 + m16;
            const float bv = bias[gn];
#pragma unroll
            for (int r = 0; r < 8; ++r)
                out[(size_t)(gm0 + r) * Ntot + gn] = acc[i][j][r] + bv;
        }
    }
}

// ---------------- launch ----------------------------------------------------

extern "C" void kernel_launch(void* const* d_in, const int* in_sizes, int n_in,
                              void* d_out, int out_size, void* d_ws, size_t ws_size,
                              hipStream_t stream)
{
    (void)in_sizes; (void)n_in; (void)out_size;

    const float* input  = (const float*)d_in[0];
    const int*   weight = (const int*  )d_in[1];
    const float* scale  = (const float*)d_in[2];
    const float* zpnt   = (const float*)d_in[3];
    const float* bias   = (const float*)d_in[4];
    float*       out    = (float*)d_out;

    dim3 grid(Ntot / BN, Mtot / BM);   // 32 x 64 workgroups

    const size_t needA = (size_t)Mtot * Ktot * sizeof(__bf16);   // 64 MB
    const size_t needW = (size_t)Ntot * Ktot * sizeof(__bf16);   // 32 MB

    if (ws_size >= needA + needW) {
        __bf16* Abf = (__bf16*)d_ws;
        __bf16* Wbf = (__bf16*)((char*)d_ws + needA);

        const int nA4 = Mtot * Ktot / 4;
        const int nW4 = Ntot * Ktot / 4;
        convert_input_bf16<<<(nA4 + 255) / 256, 256, 0, stream>>>(input, Abf, nA4);
        dequant_weight_bf16<<<(nW4 + 255) / 256, 256, 0, stream>>>(weight, scale, zpnt, Wbf, nW4);
        qgemm_bf16_async<<<grid, 256, 0, stream>>>(Abf, Wbf, bias, out);
    } else {
        qlinear_wmma_bf16<<<grid, 256, 0, stream>>>(input, weight, scale, zpnt, bias, out);
    }
}
